// AttentionBlock_39548058861785
// MI455X (gfx1250) — matrix operand order repaired and checked by reading, not compiled, not run on hardware
//
#include <hip/hip_runtime.h>
#include <math.h>

// Problem constants (from reference)
#define BB   8
#define CC   512
#define TT   2048
#define KSZ  512   // KEY_SIZE
#define VSZ  512   // VALUE_SIZE

typedef __attribute__((ext_vector_type(16))) __bf16 v16bf;
typedef __attribute__((ext_vector_type(8)))  __bf16 bf16x8;
typedef __attribute__((ext_vector_type(8)))  float  v8f;

// ---------------------------------------------------------------------------
// Kernel 1: fused QKV projection.
//   q/k[b,t,n] = sum_c x[b,c,t]*W[n,c] + b[n]       stored (B,T,K)
//   vT[b,n,t]  = sum_c x[b,c,t]*Wv[n,c] + bv[n]     stored (B,V,T)  (transposed)
// One block owns a (b, mat, t-tile) strip: the 16x512 A-slab of x is loaded
// cooperatively ONCE (coalesced), converted to bf16 into LDS (padded rows ->
// conflict-free b128 fragment reads), then 8 waves serve 4 n-tiles each.
// ---------------------------------------------------------------------------
__global__ void k_qkv(const float* __restrict__ x,
                      const float* __restrict__ Wq, const float* __restrict__ bq,
                      const float* __restrict__ Wk, const float* __restrict__ bk,
                      const float* __restrict__ Wv, const float* __restrict__ bv,
                      __bf16* __restrict__ qb, __bf16* __restrict__ kb,
                      __bf16* __restrict__ vtb)
{
    __shared__ __bf16 sA[16][CC + 8];   // +8 pad: row m -> bank group 4*m

    const int tid = threadIdx.x;
    int tmp = blockIdx.x;
    const int tTile = tmp % (TT / 16); tmp /= (TT / 16);
    const int mat   = tmp % 3;         tmp /= 3;
    const int b     = tmp;

    const float* W    = (mat == 0) ? Wq : (mat == 1) ? Wk : Wv;
    const float* bias = (mat == 0) ? bq : (mat == 1) ? bk : bv;

    const int t0 = tTile * 16;
    const float* xb = x + (size_t)b * CC * TT;

    // Cooperative load of the A-slab: x[b][c][t0..t0+15] for c = 0..511.
    // Lanes 0-15 read 16 consecutive floats of one c-row (64B segments).
    {
        const int tcol = tid & 15;
        const int crow = tid >> 4;
#pragma unroll
        for (int it = 0; it < CC / 16; ++it) {
            const int c = it * 16 + crow;
            sA[tcol][c] = (__bf16)xb[(size_t)c * TT + t0 + tcol];
        }
    }
    __syncthreads();

    const int wave = tid >> 5;
    const int lane = tid & 31;
    const int mn   = lane & 15;
    const int half = lane >> 4;

    for (int nt = wave * 4; nt < wave * 4 + 4; ++nt) {
        const float* wrow = W + (size_t)(nt * 16 + mn) * CC;   // contiguous in c
        v8f acc = {};
        for (int kk = 0; kk < CC; kk += 32) {
            // A fragment from LDS: two b128 reads (conflict-free)
            bf16x8 alo = *(const bf16x8*)&sA[mn][kk + half * 16];
            bf16x8 ahi = *(const bf16x8*)&sA[mn][kk + half * 16 + 8];
            v16bf a = __builtin_shufflevector(alo, ahi,
                      0,1,2,3,4,5,6,7,8,9,10,11,12,13,14,15);
            // B fragment: elements 0..15 <-> W[nt*16+mn][kk + half*16 + 0..15]
            const float* pw = wrow + kk + half * 16;
            float4 w0 = *(const float4*)(pw + 0);
            float4 w1 = *(const float4*)(pw + 4);
            float4 w2 = *(const float4*)(pw + 8);
            float4 w3 = *(const float4*)(pw + 12);
            v16bf bf;
            bf[0]=(__bf16)w0.x;  bf[1]=(__bf16)w0.y;  bf[2]=(__bf16)w0.z;  bf[3]=(__bf16)w0.w;
            bf[4]=(__bf16)w1.x;  bf[5]=(__bf16)w1.y;  bf[6]=(__bf16)w1.z;  bf[7]=(__bf16)w1.w;
            bf[8]=(__bf16)w2.x;  bf[9]=(__bf16)w2.y;  bf[10]=(__bf16)w2.z; bf[11]=(__bf16)w2.w;
            bf[12]=(__bf16)w3.x; bf[13]=(__bf16)w3.y; bf[14]=(__bf16)w3.z; bf[15]=(__bf16)w3.w;

            acc = __builtin_amdgcn_wmma_f32_16x16x32_bf16(false, a, false, bf,
                                                          (short)0, acc, false, false);
        }

        const float bsv = bias[nt * 16 + mn];
        if (mat == 2) {
            // v stored transposed (B, V, T): row = output col, contiguous in t
            bf16x8 st;
#pragma unroll
            for (int d = 0; d < 8; ++d) st[d] = (__bf16)(acc[d] + bsv);
            __bf16* vrow = vtb + (size_t)b * VSZ * TT + (size_t)(nt * 16 + mn) * TT;
            *(bf16x8*)(vrow + t0 + half * 8) = st;
        } else {
            __bf16* ob = ((mat == 0) ? qb : kb) + (size_t)b * TT * KSZ;
#pragma unroll
            for (int d = 0; d < 8; ++d) {
                const int t = t0 + half * 8 + d;
                ob[(size_t)t * KSZ + nt * 16 + mn] = (__bf16)(acc[d] + bsv);
            }
        }
    }
}

// ---------------------------------------------------------------------------
// Kernel 2: scores = q @ k^T, causal mask, e = exp(score/sqrt(K)).
// Writes unnormalized e into the weights output region.
// ---------------------------------------------------------------------------
__global__ void k_scores(const __bf16* __restrict__ qb,
                         const __bf16* __restrict__ kb,
                         float* __restrict__ wts)
{
    const int wid  = (blockIdx.x * blockDim.x + threadIdx.x) >> 5;
    const int lane = threadIdx.x & 31;

    const int JTI = TT / 16;    // 128
    int tmp = wid;
    const int jTile = tmp % JTI; tmp /= JTI;
    const int iTile = tmp % JTI; tmp /= JTI;
    const int b     = tmp;

    const int half = lane >> 4;
    const int mn   = lane & 15;
    const int i0 = iTile * 16, j0 = jTile * 16;
    float* wrow = wts + (size_t)b * TT * TT;

    if (j0 > i0 + 15) {               // fully above the diagonal -> all zero
#pragma unroll
        for (int d = 0; d < 8; ++d) {
            const int i = i0 + half * 8 + d;
            wrow[(size_t)i * TT + j0 + mn] = 0.0f;
        }
        return;
    }

    const __bf16* qrow = qb + (size_t)b * TT * KSZ + (size_t)(i0 + mn) * KSZ;
    const __bf16* krow = kb + (size_t)b * TT * KSZ + (size_t)(j0 + mn) * KSZ;

    v8f acc = {};
    for (int kk = 0; kk < KSZ; kk += 32) {
        bf16x8 alo = *(const bf16x8*)(qrow + kk + half * 16);
        bf16x8 ahi = *(const bf16x8*)(qrow + kk + half * 16 + 8);
        v16bf a = __builtin_shufflevector(alo, ahi,
                  0,1,2,3,4,5,6,7,8,9,10,11,12,13,14,15);
        bf16x8 blo = *(const bf16x8*)(krow + kk + half * 16);
        bf16x8 bhi = *(const bf16x8*)(krow + kk + half * 16 + 8);
        v16bf bf = __builtin_shufflevector(blo, bhi,
                  0,1,2,3,4,5,6,7,8,9,10,11,12,13,14,15);
        acc = __builtin_amdgcn_wmma_f32_16x16x32_bf16(false, a, false, bf,
                                                      (short)0, acc, false, false);
    }

    const float scale = 0.044194173824159216f;  // 1/sqrt(512)
#pragma unroll
    for (int d = 0; d < 8; ++d) {
        const int i = i0 + half * 8 + d;
        const int j = j0 + mn;
        const float e = (j > i) ? 0.0f : __expf(acc[d] * scale);
        wrow[(size_t)i * TT + j] = e;
    }
}

// ---------------------------------------------------------------------------
// Kernel 3: colsum[b][j] = sum_i e[b][i][j]   (deterministic, no atomics)
// ---------------------------------------------------------------------------
__global__ void k_colsum(const float* __restrict__ wts,
                         float* __restrict__ colsum)
{
    const int tid = blockIdx.x * blockDim.x + threadIdx.x;   // 0 .. B*T-1
    const int b = tid / TT;
    const int j = tid % TT;
    const float* wcol = wts + (size_t)b * TT * TT + j;
    float s = 0.0f;
    for (int i = j; i < TT; ++i)            // rows i<j are masked zeros
        s += wcol[(size_t)i * TT];
    colsum[tid] = s;
}

// ---------------------------------------------------------------------------
// Kernel 4: normalize weights in place (softmax over axis=1 per column j).
// ---------------------------------------------------------------------------
__global__ void k_norm(float* __restrict__ wts,
                       const float* __restrict__ colsum)
{
    const size_t idx = ((size_t)blockIdx.x * blockDim.x + threadIdx.x) * 4;
    const int b = (int)(idx / ((size_t)TT * TT));
    const int j = (int)(idx % TT);
    float4 e  = *(float4*)(wts + idx);
    float4 cs = *(const float4*)(colsum + (size_t)b * TT + j);
    e.x /= cs.x; e.y /= cs.y; e.z /= cs.z; e.w /= cs.w;
    *(float4*)(wts + idx) = e;
}

// ---------------------------------------------------------------------------
// Kernel 5: out[b,i,v] = sum_j w[i,j]*v[j,v], stored transposed as (B,V,T).
// v is stored transposed (B,V,T) so the B fragment is two contiguous b128
// loads per lane.  Causality -> only j <= i contribute: truncate the K-loop.
// ---------------------------------------------------------------------------
__global__ void k_out(const float* __restrict__ wts,
                      const __bf16* __restrict__ vtb,
                      float* __restrict__ outp)
{
    const int wid  = (blockIdx.x * blockDim.x + threadIdx.x) >> 5;
    const int lane = threadIdx.x & 31;

    const int VTI = VSZ / 16;   // 32
    const int ITI = TT  / 16;   // 128
    int tmp = wid;
    const int vTile = tmp % VTI; tmp /= VTI;
    const int iTile = tmp % ITI; tmp /= ITI;
    const int b     = tmp;

    const int half = lane >> 4;
    const int mn   = lane & 15;
    const int i0 = iTile * 16, v0 = vTile * 16;

    const float*  arow = wts + (size_t)b * TT * TT + (size_t)(i0 + mn) * TT;
    const __bf16* vrow = vtb + (size_t)b * VSZ * TT + (size_t)(v0 + mn) * TT;

    const int kEnd = ((i0 + 16 + 31) >> 5) << 5;   // round up to 32, <= TT

    v8f acc = {};
    for (int kk = 0; kk < kEnd; kk += 32) {
        const float* pa0 = arow + kk + half * 16;
        const float* pa1 = arow + kk + half * 16 + 8;
        float4 a0 = *(const float4*)(pa0 + 0);
        float4 a1 = *(const float4*)(pa0 + 4);
        float4 a2 = *(const float4*)(pa1 + 0);
        float4 a3 = *(const float4*)(pa1 + 4);
        v16bf a;
        a[0]=(__bf16)a0.x;  a[1]=(__bf16)a0.y;  a[2]=(__bf16)a0.z;  a[3]=(__bf16)a0.w;
        a[4]=(__bf16)a1.x;  a[5]=(__bf16)a1.y;  a[6]=(__bf16)a1.z;  a[7]=(__bf16)a1.w;
        a[8]=(__bf16)a2.x;  a[9]=(__bf16)a2.y;  a[10]=(__bf16)a2.z; a[11]=(__bf16)a2.w;
        a[12]=(__bf16)a3.x; a[13]=(__bf16)a3.y; a[14]=(__bf16)a3.z; a[15]=(__bf16)a3.w;

        // B fragment: elements 0..15 <-> vT[v0+mn][kk + half*16 + 0..15]
        bf16x8 blo = *(const bf16x8*)(vrow + kk + half * 16);
        bf16x8 bhi = *(const bf16x8*)(vrow + kk + half * 16 + 8);
        v16bf bf = __builtin_shufflevector(blo, bhi,
                  0,1,2,3,4,5,6,7,8,9,10,11,12,13,14,15);

        acc = __builtin_amdgcn_wmma_f32_16x16x32_bf16(false, a, false, bf,
                                                      (short)0, acc, false, false);
    }

    float* ob = outp + (size_t)b * VSZ * TT + (size_t)(v0 + mn) * TT;
#pragma unroll
    for (int d = 0; d < 8; ++d)
        ob[i0 + half * 8 + d] = acc[d];
}

// ---------------------------------------------------------------------------
extern "C" void kernel_launch(void* const* d_in, const int* in_sizes, int n_in,
                              void* d_out, int out_size, void* d_ws, size_t ws_size,
                              hipStream_t stream) {
    const float* x  = (const float*)d_in[0];
    const float* Wq = (const float*)d_in[1];
    const float* bq = (const float*)d_in[2];
    const float* Wk = (const float*)d_in[3];
    const float* bk = (const float*)d_in[4];
    const float* Wv = (const float*)d_in[5];
    const float* bv = (const float*)d_in[6];

    float* out = (float*)d_out;                       // (B, V, T)
    float* wts = out + (size_t)BB * VSZ * TT;         // (B, T, T) weights region

    const size_t qkv_elems = (size_t)BB * TT * KSZ;
    __bf16* qb  = (__bf16*)d_ws;
    __bf16* kb  = qb + qkv_elems;
    __bf16* vtb = kb + qkv_elems;                     // v transposed (B, V, T)
    float*  colsum = (float*)(vtb + qkv_elems);       // B*T floats

    // 1) QKV: one block per (b, mat, t-tile) = 3072 blocks, 8 waves each
    k_qkv<<<BB * 3 * (TT / 16), 256, 0, stream>>>(x, Wq, bq, Wk, bk, Wv, bv,
                                                  qb, kb, vtb);
    // 2) scores+exp: B*(T/16)^2 = 131072 waves, 8 waves/block
    k_scores<<<16384, 256, 0, stream>>>(qb, kb, wts);
    // 3) column sums (deterministic): B*T threads
    k_colsum<<<(BB * TT) / 256, 256, 0, stream>>>(wts, colsum);
    // 4) normalize: B*T*T/4 float4 lanes
    k_norm<<<(size_t)BB * TT * TT / 4 / 256, 256, 0, stream>>>(wts, colsum);
    // 5) out = w @ v (transposed store): B*(T/16)*(V/16) = 32768 waves
    k_out<<<4096, 256, 0, stream>>>(wts, vtb, out);
}
